// Ferminet_70635032150790
// MI455X (gfx1250) — matrix operand order repaired and checked by reading, not compiled, not run on hardware
//
#include <hip/hip_runtime.h>

typedef unsigned short ushort_t;
typedef __attribute__((ext_vector_type(16))) __bf16    v16bf;
typedef __attribute__((ext_vector_type(8)))  float     v8f;
typedef __attribute__((ext_vector_type(4)))  unsigned  uint4e;

union BFrag {
    uint4e  q[2];
    unsigned u[8];
    v16bf   v;
};

__device__ __forceinline__ ushort_t f2bf(float x) {
    unsigned u = __builtin_bit_cast(unsigned, x);
    unsigned r = u + 0x7FFFu + ((u >> 16) & 1u);   // RNE to bf16
    return (ushort_t)(r >> 16);
}
__device__ __forceinline__ float bf2f(ushort_t h) {
    unsigned u = ((unsigned)h) << 16;
    return __builtin_bit_cast(float, u);
}
__device__ __forceinline__ unsigned pack2(float a, float b) {
    return (unsigned)f2bf(a) | ((unsigned)f2bf(b) << 16);
}

// ---------------- workspace layout (in bf16 halves) ----------------
// vw0:  Kpad=64 ,N=256 -> 16 ntiles * 2 kt * 512 = 16384
// vw1-3:Kpad=832,N=256 -> 16 * 26 * 512 = 212992 each
// ww0-3:Kpad=32 ,N=32  -> 2 * 1 * 512 = 1024 each
// env_w A-operand: 2 spin * 8 mtile * 8 ktile * 512 = 65536
#define OFF_VW0 0
#define OFF_VW1 16384
#define OFF_VW2 (16384 + 212992)
#define OFF_VW3 (16384 + 2 * 212992)
#define OFF_WW  (16384 + 3 * 212992)
#define OFF_ENV (OFF_WW + 4 * 1024)
#define WS_HALVES (OFF_ENV + 65536)   // 724992 halves = ~1.45 MB

// Pre-swizzle a (Ksrc x N) fp32 weight into WMMA B-operand blocks
// [ntile][ktile][lane(32)][8 dwords]; lane n<16 = col n, K 0..7,16..23 of tile;
// lane n>=16 = col n-16, K 8..15,24..31.  Zero-pad K to Kpad = KT*32.
__global__ void swizzleB_kernel(const float* __restrict__ src, ushort_t* __restrict__ dst,
                                int Ksrc, int KT, int NT) {
    int N = NT * 16;
    int total = NT * KT * 256;    // dwords
    for (int idx = blockIdx.x * blockDim.x + threadIdx.x; idx < total;
         idx += gridDim.x * blockDim.x) {
        int bid = idx >> 8, within = idx & 255;
        int n = within >> 3, v = within & 7;
        int nt = bid / KT, kt = bid - nt * KT;
        int col = nt * 16 + (n & 15);
        int k0  = kt * 32 + ((n & 16) ? 8 : 0) + ((v < 4) ? 2 * v : 16 + 2 * (v - 4));
        float a = (k0     < Ksrc) ? src[k0 * N + col]       : 0.f;
        float b = (k0 + 1 < Ksrc) ? src[(k0 + 1) * N + col] : 0.f;
        ((unsigned*)dst)[idx] = pack2(a, b);
    }
}

// Pre-swizzle env_w (DET=16, E=16, D1=256) into WMMA A-operand blocks
// [spin][mtile(8)][ktile(8)][lane][8 dwords]; lane L: row M = L&15,
// K-range base +8 for lanes>=16; v0..3 = K b..b+7, v4..7 = K b+16..b+23.
__global__ void swizzleEnv_kernel(const float* __restrict__ ew, ushort_t* __restrict__ dst) {
    const int total = 32768;  // dwords
    for (int idx = blockIdx.x * blockDim.x + threadIdx.x; idx < total;
         idx += gridDim.x * blockDim.x) {
        int blk = idx >> 8, within = idx & 255;
        int L = within >> 3, v = within & 7;
        int sp = blk >> 6, mt = (blk >> 3) & 7, kt = blk & 7;
        int m = mt * 16 + (L & 15);
        int kdet = m >> 3, i = m & 7, e = sp * 8 + i;
        unsigned outw = 0;
        #pragma unroll
        for (int h = 0; h < 2; ++h) {
            int kk = (v < 4) ? (2 * v + h) : (16 + 2 * (v - 4) + h);
            int K = kt * 32 + ((L >= 16) ? 8 : 0) + kk;
            outw |= ((unsigned)f2bf(ew[(kdet * 16 + e) * 256 + K])) << (16 * h);
        }
        ((unsigned*)dst)[idx] = outw;
    }
}

#define WMMA_BF16(A, B, C) \
    __builtin_amdgcn_wmma_f32_16x16x32_bf16(false, (A), false, (B), (short)0, (C), false, false)

__global__ __launch_bounds__(256) void ferminet_fused(
    const float* __restrict__ x,    const float* __restrict__ nucp,
    const float* __restrict__ vb0,  const float* __restrict__ vb1,
    const float* __restrict__ vb2,  const float* __restrict__ vb3,
    const float* __restrict__ wb0,  const float* __restrict__ wb1,
    const float* __restrict__ wb2,  const float* __restrict__ wb3,
    const float* __restrict__ env_g, const float* __restrict__ sigma,
    const float* __restrict__ pivals, const ushort_t* __restrict__ ws,
    float* __restrict__ out)
{
    __shared__ float s_h1[16 * 256];                    // fp32 master h1
    __shared__ alignas(16) ushort_t s_f[16 * 832];      // bf16 f (A operand); psi aliases later
    __shared__ alignas(16) ushort_t s_h2b[256 * 32];    // bf16 master h2
    __shared__ float s_d1[16 * 4];
    __shared__ float s_pos[48];

    const int tid  = threadIdx.x;
    const int lane = tid & 31;
    const int w    = tid >> 5;
    const int b    = blockIdx.x;

    // ---------------- setup: features ----------------
    if (tid < 48) s_pos[tid] = x[b * 48 + tid];
    __syncthreads();
    {   // pair features, one (i,j) per thread
        int i = tid >> 4, j = tid & 15;
        float dx = s_pos[j * 3 + 0] - s_pos[i * 3 + 0];
        float dy = s_pos[j * 3 + 1] - s_pos[i * 3 + 1];
        float dz = s_pos[j * 3 + 2] - s_pos[i * 3 + 2];
        float dd = sqrtf(dx * dx + dy * dy + dz * dz);
        ushort_t* row = s_h2b + tid * 32;
        row[0] = f2bf(dx); row[1] = f2bf(dy); row[2] = f2bf(dz); row[3] = f2bf(dd);
        #pragma unroll
        for (int c = 4; c < 32; ++c) row[c] = 0;        // K padding 4->32
    }
    if (tid < 16) {
        int e = tid;
        #pragma unroll
        for (int n = 0; n < 4; ++n) {
            float rx = s_pos[e * 3 + 0] - nucp[n * 3 + 0];
            float ry = s_pos[e * 3 + 1] - nucp[n * 3 + 1];
            float rz = s_pos[e * 3 + 2] - nucp[n * 3 + 2];
            float dd = sqrtf(rx * rx + ry * ry + rz * rz);
            s_h1[e * 256 + n * 4 + 0] = rx;
            s_h1[e * 256 + n * 4 + 1] = ry;
            s_h1[e * 256 + n * 4 + 2] = rz;
            s_h1[e * 256 + n * 4 + 3] = dd;
            s_d1[e * 4 + n] = dd;
        }
    }
    __syncthreads();

    const float* vbs[4] = {vb0, vb1, vb2, vb3};
    const float* wbs[4] = {wb0, wb1, wb2, wb3};

    // ---------------- layer loop ----------------
    for (int l = 0; l < 4; ++l) {
        const int C1 = (l == 0) ? 16 : 256;
        const int C2 = (l == 0) ? 4 : 32;
        const int FK = 3 * C1 + 2 * C2;        // 56 or 832
        const int FP = (l == 0) ? 64 : 832;    // K padded to 32-multiple
        const int KT = FP >> 5;

        // build f = [h1 | g1u | g1d | g2u | g2d | 0-pad] as bf16
        for (int idx = tid; idx < 16 * FP; idx += 256) {
            int i = idx / FP, c = idx - i * FP;
            float val = 0.f;
            if (c < C1) {
                val = s_h1[i * 256 + c];
            } else if (c < 2 * C1) {
                int cc = c - C1; float s = 0.f;
                #pragma unroll
                for (int e = 0; e < 8; ++e) s += s_h1[e * 256 + cc];
                val = s * 0.125f;
            } else if (c < 3 * C1) {
                int cc = c - 2 * C1; float s = 0.f;
                #pragma unroll
                for (int e = 8; e < 16; ++e) s += s_h1[e * 256 + cc];
                val = s * 0.125f;
            } else if (c < 3 * C1 + C2) {
                int cc = c - 3 * C1; float s = 0.f;
                #pragma unroll
                for (int jj = 0; jj < 8; ++jj) s += bf2f(s_h2b[(i * 16 + jj) * 32 + cc]);
                val = s * 0.125f;
            } else if (c < FK) {
                int cc = c - 3 * C1 - C2; float s = 0.f;
                #pragma unroll
                for (int jj = 8; jj < 16; ++jj) s += bf2f(s_h2b[(i * 16 + jj) * 32 + cc]);
                val = s * 0.125f;
            }
            s_f[idx] = f2bf(val);
        }
        __syncthreads();

        // ---- GEMM2 A operands (snapshot h2 rows before overwrite) ----
        const int m0 = 2 * w, m1 = m0 + 1;
        BFrag A20, A21;
        {
            int kb = (lane & 16) ? 8 : 0;
            const uint4e* p0 = (const uint4e*)(s_h2b + (m0 * 16 + (lane & 15)) * 32 + kb);
            A20.q[0] = p0[0]; A20.q[1] = p0[2];
            const uint4e* p1 = (const uint4e*)(s_h2b + (m1 * 16 + (lane & 15)) * 32 + kb);
            A21.q[0] = p1[0]; A21.q[1] = p1[2];
        }

        // ---- GEMM1: nh1 = tanh(f @ vw + vb) (+h1)  M=16,N=256,K=FP ----
        v8f acc0 = {}, acc1 = {};
        const ushort_t* wv = ws + ((l == 0) ? OFF_VW0 : (OFF_VW1 + (l - 1) * 212992));
        const int n0 = 2 * w, n1 = n0 + 1;
        for (int kt = 0; kt < KT; ++kt) {
            BFrag Af;
            {
                int kb = kt * 32 + ((lane & 16) ? 8 : 0);
                const uint4e* ap = (const uint4e*)(s_f + (lane & 15) * FP + kb);
                Af.q[0] = ap[0]; Af.q[1] = ap[2];
            }
            BFrag B0, B1;
            const uint4e* b0 = (const uint4e*)(wv + (size_t)(n0 * KT + kt) * 512) + lane * 2;
            B0.q[0] = b0[0]; B0.q[1] = b0[1];
            const uint4e* b1 = (const uint4e*)(wv + (size_t)(n1 * KT + kt) * 512) + lane * 2;
            B1.q[0] = b1[0]; B1.q[1] = b1[1];
            acc0 = WMMA_BF16(Af.v, B0.v, acc0);
            acc1 = WMMA_BF16(Af.v, B1.v, acc1);
        }

        // ---- GEMM2: nh2 = tanh(h2 @ ww + wb) (+h2)  M=256,N=32,K=32 ----
        v8f c00 = {}, c01 = {}, c10 = {}, c11 = {};
        {
            const ushort_t* wwp = ws + OFF_WW + l * 1024;
            BFrag Bw0, Bw1;
            const uint4e* p0 = (const uint4e*)wwp + lane * 2;
            Bw0.q[0] = p0[0]; Bw0.q[1] = p0[1];
            const uint4e* p1 = (const uint4e*)(wwp + 512) + lane * 2;
            Bw1.q[0] = p1[0]; Bw1.q[1] = p1[1];
            c00 = WMMA_BF16(A20.v, Bw0.v, c00);
            c01 = WMMA_BF16(A20.v, Bw1.v, c01);
            c10 = WMMA_BF16(A21.v, Bw0.v, c10);
            c11 = WMMA_BF16(A21.v, Bw1.v, c11);
        }

        // ---- epilogues ----
        const float* vb = vbs[l];
        const float* wb = wbs[l];
        const int mbase = (lane >> 4) << 3;
        const int nc0 = n0 * 16 + (lane & 15);
        const int nc1 = n1 * 16 + (lane & 15);
        #pragma unroll
        for (int r = 0; r < 8; ++r) {
            int m = r + mbase;
            float xv = tanhf(acc0[r] + vb[nc0]);
            if (l > 0) xv += s_h1[m * 256 + nc0];
            s_h1[m * 256 + nc0] = xv;
            float yv = tanhf(acc1[r] + vb[nc1]);
            if (l > 0) yv += s_h1[m * 256 + nc1];
            s_h1[m * 256 + nc1] = yv;
        }
        #pragma unroll
        for (int r = 0; r < 8; ++r) {
            int m = r + mbase;
            int row0 = m0 * 16 + m, row1 = m1 * 16 + m;
            int col0 = lane & 15, col1 = 16 + (lane & 15);
            float t00 = tanhf(c00[r] + wb[col0]); if (l > 0) t00 += bf2f(s_h2b[row0 * 32 + col0]);
            s_h2b[row0 * 32 + col0] = f2bf(t00);
            float t01 = tanhf(c01[r] + wb[col1]); if (l > 0) t01 += bf2f(s_h2b[row0 * 32 + col1]);
            s_h2b[row0 * 32 + col1] = f2bf(t01);
            float t10 = tanhf(c10[r] + wb[col0]); if (l > 0) t10 += bf2f(s_h2b[row1 * 32 + col0]);
            s_h2b[row1 * 32 + col0] = f2bf(t10);
            float t11 = tanhf(c11[r] + wb[col1]); if (l > 0) t11 += bf2f(s_h2b[row1 * 32 + col1]);
            s_h2b[row1 * 32 + col1] = f2bf(t11);
        }
        __syncthreads();
    }

    // ---------------- orbitals: psi[k,i,j] = (W h1s^T + 256 g) * env ----------------
    float* s_psi  = (float*)s_f;          // s_f is dead now; 2048 floats
    float* s_detv = s_psi + 2048;         // 32 floats
    for (int q = 0; q < 2; ++q) {
        int u = 2 * w + q;                // 16 units: spin x 8 mtiles
        int sp = u >> 3, mt = u & 7;
        v8f acc = {};
        for (int kt = 0; kt < 8; ++kt) {
            BFrag Ae;
            const uint4e* ap =
                (const uint4e*)(ws + OFF_ENV + (size_t)((sp * 8 + mt) * 8 + kt) * 512) + lane * 2;
            Ae.q[0] = ap[0]; Ae.q[1] = ap[1];
            BFrag Bh;
            int el = lane & 15;
            if (el < 8) {   // B col n = h1 row (spin_base+n); cols 8..15 are zero
                const float* hp = s_h1 + (sp * 8 + el) * 256 + kt * 32 + ((lane & 16) ? 8 : 0);
                #pragma unroll
                for (int v = 0; v < 8; ++v) { int kk = (v < 4) ? 2 * v : 16 + 2 * (v - 4); Bh.u[v] = pack2(hp[kk], hp[kk + 1]); }
            } else {
                #pragma unroll
                for (int v = 0; v < 8; ++v) Bh.u[v] = 0u;
            }
            acc = WMMA_BF16(Ae.v, Bh.v, acc);
        }
        int j = lane & 15;
        if (j < 8) {
            #pragma unroll
            for (int r = 0; r < 8; ++r) {
                int m = mt * 16 + r + ((lane >> 4) << 3);
                int kd = m >> 3, i = m & 7;
                int gi = kd * 16 + sp * 8 + i;
                float dotv = acc[r] + 256.0f * env_g[gi];
                float env = 0.f;
                #pragma unroll
                for (int n = 0; n < 4; ++n) {
                    float sg = fabsf(sigma[gi * 4 + n]);
                    env += pivals[gi * 4 + n] * __expf(-sg * s_d1[(sp * 8 + j) * 4 + n]);
                }
                s_psi[(sp * 16 + kd) * 64 + i * 8 + j] = dotv * env;
            }
        }
    }
    __syncthreads();

    // ---------------- 8x8 determinants (wave 0, one matrix per lane) ----------------
    if (tid < 32) {
        int sp = tid >> 4, kd = tid & 15;
        float M[8][8];
        #pragma unroll
        for (int i = 0; i < 8; ++i)
            #pragma unroll
            for (int j = 0; j < 8; ++j)
                M[i][j] = s_psi[(sp * 16 + kd) * 64 + i * 8 + j];
        float det = 1.f;
        #pragma unroll
        for (int p = 0; p < 8; ++p) {
            float d = M[p][p];
            det *= d;
            float inv = 1.0f / d;
            #pragma unroll
            for (int r = p + 1; r < 8; ++r) {
                float fct = M[r][p] * inv;
                #pragma unroll
                for (int c = p; c < 8; ++c) M[r][c] -= fct * M[p][c];
            }
        }
        s_detv[tid] = det;
    }
    __syncthreads();
    if (tid == 0) {
        float sum = 0.f;
        #pragma unroll
        for (int k = 0; k < 16; ++k) sum += s_detv[k] * s_detv[16 + k];
        out[b] = sum;
    }
}

extern "C" void kernel_launch(void* const* d_in, const int* in_sizes, int n_in,
                              void* d_out, int out_size, void* d_ws, size_t ws_size,
                              hipStream_t stream) {
    (void)in_sizes; (void)n_in; (void)out_size; (void)ws_size;
    // dict order: x, nuc, vw[0..3], vb[0..3], ww[0..3], wb[0..3], env_w, env_g, sigma, pi
    const float* x    = (const float*)d_in[0];
    const float* nuc  = (const float*)d_in[1];
    const float* vw[4] = {(const float*)d_in[2], (const float*)d_in[3],
                          (const float*)d_in[4], (const float*)d_in[5]};
    const float* vb[4] = {(const float*)d_in[6], (const float*)d_in[7],
                          (const float*)d_in[8], (const float*)d_in[9]};
    const float* ww[4] = {(const float*)d_in[10], (const float*)d_in[11],
                          (const float*)d_in[12], (const float*)d_in[13]};
    const float* wb[4] = {(const float*)d_in[14], (const float*)d_in[15],
                          (const float*)d_in[16], (const float*)d_in[17]};
    const float* env_w = (const float*)d_in[18];
    const float* env_g = (const float*)d_in[19];
    const float* sig   = (const float*)d_in[20];
    const float* piv   = (const float*)d_in[21];
    ushort_t* ws = (ushort_t*)d_ws;
    float* out = (float*)d_out;

    // per-call weight pre-swizzle into WMMA operand layouts (cheap, deterministic)
    swizzleB_kernel<<<32, 256, 0, stream>>>(vw[0], ws + OFF_VW0, 56, 2, 16);
    swizzleB_kernel<<<416, 256, 0, stream>>>(vw[1], ws + OFF_VW1, 832, 26, 16);
    swizzleB_kernel<<<416, 256, 0, stream>>>(vw[2], ws + OFF_VW2, 832, 26, 16);
    swizzleB_kernel<<<416, 256, 0, stream>>>(vw[3], ws + OFF_VW3, 832, 26, 16);
    for (int l = 0; l < 4; ++l)
        swizzleB_kernel<<<2, 256, 0, stream>>>(ww[l], ws + OFF_WW + l * 1024,
                                               (l == 0) ? 4 : 32, 1, 2);
    swizzleEnv_kernel<<<128, 256, 0, stream>>>(env_w, ws + OFF_ENV);

    ferminet_fused<<<4096, 256, 0, stream>>>(
        x, nuc, vb[0], vb[1], vb[2], vb[3], wb[0], wb[1], wb[2], wb[3],
        env_g, sig, piv, ws, out);
}
